// FeedForwardTransformer_43508018709268
// MI455X (gfx1250) — matrix operand order repaired and checked by reading, not compiled, not run on hardware
//
#include <hip/hip_runtime.h>

// ---------------------------------------------------------------------------
// Transformer block (attention + proj + LN + conv1d + LN) for gfx1250 (MI455X)
// bf16 WMMA (v_wmma_f32_16x16x32_bf16) for every GEMM-like op. All weights are
// pre-transposed to N-major bf16 so every A and B fragment is built from
// contiguous dword loads (lowered to global_load_b128 pairs). Flash-attention
// online softmax avoids the 1GB score tensor. 32x64 GEMM wave tiles
// (8 WMMA/K-step); all fragment loads issued before the WMMA burst so the
// scheduler can clause them and stagger the loadcnt waits.
// ---------------------------------------------------------------------------

#define B_ 4
#define S_ 2048
#define C_ 1024
#define H_ 16
#define DH_ 64
#define BS_ (B_ * S_)      // 8192 tokens

typedef __attribute__((ext_vector_type(16))) __bf16 v16bf;
typedef __attribute__((ext_vector_type(8)))  float  v8f;

union FragB16 {
    v16bf v;
    unsigned short h[16];
    unsigned int   w[8];
};

__device__ __forceinline__ v8f zero8() {
    v8f z = {0.f, 0.f, 0.f, 0.f, 0.f, 0.f, 0.f, 0.f};
    return z;
}

__device__ __forceinline__ unsigned short f2bf(float x) {
    unsigned int u = __float_as_uint(x);
    u += 0x7FFFu + ((u >> 16) & 1u);   // round-to-nearest-even
    return (unsigned short)(u >> 16);
}

__device__ __forceinline__ v8f wmma_bf16(v16bf a, v16bf b, v8f c) {
    return __builtin_amdgcn_wmma_f32_16x16x32_bf16(
        /*neg_a=*/false, a, /*neg_b=*/false, b,
        /*c_mod=*/(short)0, c, /*reuse_a=*/false, /*reuse_b=*/false);
}

// A-fragment (16x32 bf16, row-major M x K source, K pairs contiguous).
// ISA layout: lanes 0-15 row M=lane hold K {0..7,16..23}; lanes 16-31 hold
// K {8..15,24..31}. VGPR v pair = K (v>>2)*16 + kb + 2*(v&3).
__device__ __forceinline__ v16bf load_a_frag(const unsigned short* A, int lda, int lane) {
    FragB16 f;
    const int m  = lane & 15;
    const int kb = (lane & 16) ? 8 : 0;
    const unsigned short* row = A + m * lda;
#pragma unroll
    for (int r = 0; r < 8; ++r) {
        const int k = ((r >> 2) << 4) + kb + 2 * (r & 3);
        f.w[r] = *reinterpret_cast<const unsigned int*>(row + k);
    }
    return f.v;
}

// B-fragment (32x16 bf16) from an N-major (transposed, [N][K] row-major)
// source: lane holds column N=lane&15; lanes 0-15 K {0..7,16..23}, lanes 16-31
// K {8..15,24..31} (matches A); VGPR v pair = K (v>>2)*16 + kb + 2*(v&3).
// Bt points at (n0, k0).
__device__ __forceinline__ v16bf load_bT_frag(const unsigned short* Bt, int ldk, int lane) {
    FragB16 f;
    const int n  = lane & 15;
    const int kb = (lane & 16) ? 8 : 0;
    const unsigned short* row = Bt + n * ldk + kb;
#pragma unroll
    for (int r = 0; r < 8; ++r)
        f.w[r] = *reinterpret_cast<const unsigned int*>(row + ((r >> 2) << 4) + 2 * (r & 3));
    return f.v;
}

// ---------------------------------------------------------------------------
// Conversion / transposition pre-passes (bandwidth-trivial, one-time)
// ---------------------------------------------------------------------------
__global__ void convert_f32_bf16(const float* __restrict__ src,
                                 unsigned short* __restrict__ dst, int n) {
    int i = blockIdx.x * blockDim.x + threadIdx.x;
    if (i < n) dst[i] = f2bf(src[i]);
}

// src [K x N] row-major f32 -> dst [N x K] row-major bf16
__global__ void transpose_f32_bf16(const float* __restrict__ src,
                                   unsigned short* __restrict__ dst,
                                   int K, int N) {
    int i = blockIdx.x * blockDim.x + threadIdx.x;
    if (i >= N * K) return;
    const int n = i / K, k = i - n * K;
    dst[i] = f2bf(src[(size_t)k * N + n]);
}

// Wconv [co][ci][3] f32 -> dst [tap][co][ci] bf16
__global__ void conv_w_transpose(const float* __restrict__ src,
                                 unsigned short* __restrict__ dst) {
    int i = blockIdx.x * blockDim.x + threadIdx.x;
    if (i >= 3 * C_ * C_) return;
    const int tap = i / (C_ * C_);
    const int rem = i - tap * (C_ * C_);
    const int co = rem / C_, ci = rem - co * C_;
    dst[i] = f2bf(src[((size_t)co * C_ + ci) * 3 + tap]);
}

// ---------------------------------------------------------------------------
// QKV GEMM: [8192 x 1024] @ WkqvT + bias. 32x64 wave tiles. Scatters
// q,k bf16 as [b][h][s][d] and v as V^T [b][h][d][s] for the attention pass.
// ---------------------------------------------------------------------------
__global__ __launch_bounds__(256) void qkv_gemm_kernel(
    const unsigned short* __restrict__ A,    // inp bf16  [8192 x 1024]
    const unsigned short* __restrict__ Wt,   // WkqvT bf16 [3072 x 1024]
    const float* __restrict__ bias,          // bkqv [3072]
    unsigned short* __restrict__ qb,         // [b][h][s][d]
    unsigned short* __restrict__ kb,         // [b][h][s][d]
    unsigned short* __restrict__ vbT) {      // [b][h][d][s]
    const int lane = threadIdx.x & 31;
    const int wave = blockIdx.x * (blockDim.x >> 5) + (threadIdx.x >> 5);
    const int NT = 3 * C_ / 64;              // 48 column tiles
    const int mt = wave / NT, nt = wave % NT;
    const int m0 = mt * 32, n0 = nt * 64;

    v8f acc[2][4];
#pragma unroll
    for (int mh = 0; mh < 2; ++mh)
#pragma unroll
        for (int sub = 0; sub < 4; ++sub) acc[mh][sub] = zero8();

    for (int k0 = 0; k0 < C_; k0 += 32) {
        if (k0 + 32 < C_) {
            __builtin_prefetch(A + (size_t)m0 * C_ + k0 + 32, 0, 1);
            __builtin_prefetch(Wt + (size_t)n0 * C_ + k0 + 32, 0, 1);
        }
        // issue all 6 fragment loads up front, then the WMMA burst
        const v16bf a0 = load_a_frag(A + (size_t)m0 * C_ + k0, C_, lane);
        const v16bf a1 = load_a_frag(A + (size_t)(m0 + 16) * C_ + k0, C_, lane);
        v16bf b[4];
#pragma unroll
        for (int sub = 0; sub < 4; ++sub)
            b[sub] = load_bT_frag(Wt + (size_t)(n0 + sub * 16) * C_ + k0, C_, lane);
#pragma unroll
        for (int sub = 0; sub < 4; ++sub) {
            acc[0][sub] = wmma_bf16(a0, b[sub], acc[0][sub]);
            acc[1][sub] = wmma_bf16(a1, b[sub], acc[1][sub]);
        }
    }

    // 64-wide column tile lies in a single (head, q/k/v) slab (wave-uniform)
    const int h     = n0 / (3 * DH_);
    const int which = (n0 % (3 * DH_)) / DH_;
    const int half = (lane & 16) ? 8 : 0;
    const int nloc = lane & 15;
#pragma unroll
    for (int sub = 0; sub < 4; ++sub) {
        const int d  = sub * 16 + nloc;
        const float bv = bias[n0 + sub * 16 + nloc];
#pragma unroll
        for (int mh = 0; mh < 2; ++mh) {
#pragma unroll
            for (int r = 0; r < 8; ++r) {
                const int m = m0 + mh * 16 + r + half;
                const int b_idx = m >> 11;
                const int s = m & (S_ - 1);
                const unsigned short val = f2bf(acc[mh][sub][r] + bv);
                if (which == 0)
                    qb[(((size_t)b_idx * H_ + h) * S_ + s) * DH_ + d] = val;
                else if (which == 1)
                    kb[(((size_t)b_idx * H_ + h) * S_ + s) * DH_ + d] = val;
                else
                    vbT[(((size_t)b_idx * H_ + h) * DH_ + d) * S_ + s] = val;
            }
        }
    }
}

// ---------------------------------------------------------------------------
// Flash attention: one wave per 16-query block per (b,h). Online softmax,
// P transposed into A-fragment layout through a per-wave LDS tile; V consumed
// through V^T so all fragment loads are contiguous dwords.
// ---------------------------------------------------------------------------
__global__ __launch_bounds__(256) void attn_kernel(
    const unsigned short* __restrict__ q16,
    const unsigned short* __restrict__ k16,
    const unsigned short* __restrict__ vT16,
    unsigned short* __restrict__ o16) {
    __shared__ unsigned short pbuf[8][16 * 32];   // per-wave P tile (1KB each)

    const int lane  = threadIdx.x & 31;
    const int wslot = threadIdx.x >> 5;
    const int wid   = blockIdx.x * 8 + wslot;     // 0..8191
    const int bh    = wid >> 7;                   // b*16+h
    const int sblk  = wid & 127;
    const int m0    = sblk * 16;

    const unsigned short* Q  = q16  + (size_t)bh * S_ * DH_;
    const unsigned short* K  = k16  + (size_t)bh * S_ * DH_;
    const unsigned short* VT = vT16 + (size_t)bh * DH_ * S_;
    unsigned short* P = pbuf[wslot];

    const v16bf aq0 = load_a_frag(Q + m0 * DH_ + 0,  DH_, lane);
    const v16bf aq1 = load_a_frag(Q + m0 * DH_ + 32, DH_, lane);

    float mrow[8], lrow[8];
    v8f o[4];
#pragma unroll
    for (int r = 0; r < 8; ++r) { mrow[r] = -1e30f; lrow[r] = 0.f; }
#pragma unroll
    for (int i = 0; i < 4; ++i) o[i] = zero8();

    const int half = (lane & 16) ? 8 : 0;
    const int nloc = lane & 15;

    for (int t0 = 0; t0 < S_; t0 += 32) {
        // scores: K^T is B with N=t rows in memory (k16 is [t][d]) -> dwords.
        // Load all 4 K fragments first, then the WMMA burst.
        v16bf bk[4];
        bk[0] = load_bT_frag(K + (size_t)t0 * DH_ + 0,  DH_, lane);
        bk[1] = load_bT_frag(K + (size_t)t0 * DH_ + 32, DH_, lane);
        bk[2] = load_bT_frag(K + (size_t)(t0 + 16) * DH_ + 0,  DH_, lane);
        bk[3] = load_bT_frag(K + (size_t)(t0 + 16) * DH_ + 32, DH_, lane);
        v8f s0 = zero8(), s1 = zero8();
        s0 = wmma_bf16(aq0, bk[0], s0);
        s0 = wmma_bf16(aq1, bk[1], s0);
        s1 = wmma_bf16(aq0, bk[2], s1);
        s1 = wmma_bf16(aq1, bk[3], s1);

        // prefetch V fragments for this key block while doing softmax
        v16bf bv[4];
#pragma unroll
        for (int dt = 0; dt < 4; ++dt)
            bv[dt] = load_bT_frag(VT + (size_t)(dt * 16) * S_ + t0, S_, lane);

#pragma unroll
        for (int r = 0; r < 8; ++r) {
            const float a = s0[r] * 0.125f;      // 1/sqrt(64)
            const float b = s1[r] * 0.125f;
            float mx = fmaxf(a, b);
            mx = fmaxf(mx, __shfl_xor(mx, 1, 32));
            mx = fmaxf(mx, __shfl_xor(mx, 2, 32));
            mx = fmaxf(mx, __shfl_xor(mx, 4, 32));
            mx = fmaxf(mx, __shfl_xor(mx, 8, 32));   // row max within 16-lane half
            const float mnew  = fmaxf(mrow[r], mx);
            const float alpha = __expf(mrow[r] - mnew);
            const float p0 = __expf(a - mnew);
            const float p1 = __expf(b - mnew);
            float ps = p0 + p1;
            ps += __shfl_xor(ps, 1, 32);
            ps += __shfl_xor(ps, 2, 32);
            ps += __shfl_xor(ps, 4, 32);
            ps += __shfl_xor(ps, 8, 32);
            lrow[r] = lrow[r] * alpha + ps;
            mrow[r] = mnew;
            o[0][r] *= alpha; o[1][r] *= alpha; o[2][r] *= alpha; o[3][r] *= alpha;
            const int prow = r + half;
            P[prow * 32 + nloc]      = f2bf(p0);
            P[prow * 32 + 16 + nloc] = f2bf(p1);
        }

        const v16bf ap = load_a_frag(P, 32, lane);   // P in A layout via LDS
#pragma unroll
        for (int dt = 0; dt < 4; ++dt)
            o[dt] = wmma_bf16(ap, bv[dt], o[dt]);
    }

    const int b_idx = bh >> 4, h = bh & 15;
#pragma unroll
    for (int dt = 0; dt < 4; ++dt) {
        const int d = dt * 16 + nloc;
#pragma unroll
        for (int r = 0; r < 8; ++r) {
            const int srow = m0 + r + half;
            o16[((size_t)(b_idx * S_ + srow)) * C_ + h * DH_ + d] =
                f2bf(o[dt][r] / lrow[r]);
        }
    }
}

// ---------------------------------------------------------------------------
// Proj GEMM: o @ Wproj + bproj + inp -> x (f32). 32x64 wave tiles.
// ---------------------------------------------------------------------------
__global__ __launch_bounds__(256) void proj_gemm_kernel(
    const unsigned short* __restrict__ A,   // o bf16 [8192 x 1024]
    const unsigned short* __restrict__ Wt,  // WprojT bf16 [1024 x 1024]
    const float* __restrict__ bias,
    const float* __restrict__ resid,        // original inp (f32)
    float* __restrict__ X) {
    const int lane = threadIdx.x & 31;
    const int wave = blockIdx.x * (blockDim.x >> 5) + (threadIdx.x >> 5);
    const int NT = C_ / 64;                 // 16
    const int mt = wave / NT, nt = wave % NT;
    const int m0 = mt * 32, n0 = nt * 64;

    v8f acc[2][4];
#pragma unroll
    for (int mh = 0; mh < 2; ++mh)
#pragma unroll
        for (int sub = 0; sub < 4; ++sub) acc[mh][sub] = zero8();

    for (int k0 = 0; k0 < C_; k0 += 32) {
        if (k0 + 32 < C_) {
            __builtin_prefetch(A + (size_t)m0 * C_ + k0 + 32, 0, 1);
            __builtin_prefetch(Wt + (size_t)n0 * C_ + k0 + 32, 0, 1);
        }
        const v16bf a0 = load_a_frag(A + (size_t)m0 * C_ + k0, C_, lane);
        const v16bf a1 = load_a_frag(A + (size_t)(m0 + 16) * C_ + k0, C_, lane);
        v16bf b[4];
#pragma unroll
        for (int sub = 0; sub < 4; ++sub)
            b[sub] = load_bT_frag(Wt + (size_t)(n0 + sub * 16) * C_ + k0, C_, lane);
#pragma unroll
        for (int sub = 0; sub < 4; ++sub) {
            acc[0][sub] = wmma_bf16(a0, b[sub], acc[0][sub]);
            acc[1][sub] = wmma_bf16(a1, b[sub], acc[1][sub]);
        }
    }

    const int half = (lane & 16) ? 8 : 0;
#pragma unroll
    for (int sub = 0; sub < 4; ++sub) {
        const int n = n0 + sub * 16 + (lane & 15);
        const float bv = bias[n];
#pragma unroll
        for (int mh = 0; mh < 2; ++mh)
#pragma unroll
            for (int r = 0; r < 8; ++r) {
                const size_t idx = (size_t)(m0 + mh * 16 + r + half) * C_ + n;
                X[idx] = acc[mh][sub][r] + bv + resid[idx];
            }
    }
}

// ---------------------------------------------------------------------------
// LayerNorm: one block per row of 1024. Optionally also emits bf16 copy.
// ---------------------------------------------------------------------------
__global__ __launch_bounds__(256) void layernorm_kernel(
    const float* __restrict__ X,
    const float* __restrict__ g,
    const float* __restrict__ b,
    float* __restrict__ Yf,
    unsigned short* __restrict__ Y16) {     // may be nullptr
    __shared__ float rs[256], rss[256];
    const int tid = threadIdx.x;
    const float* x = X + (size_t)blockIdx.x * C_;

    float s = 0.f, ss = 0.f;
#pragma unroll
    for (int i = tid; i < C_; i += 256) { const float v = x[i]; s += v; ss += v * v; }
    rs[tid] = s; rss[tid] = ss;
    __syncthreads();
    for (int off = 128; off > 0; off >>= 1) {
        if (tid < off) { rs[tid] += rs[tid + off]; rss[tid] += rss[tid + off]; }
        __syncthreads();
    }
    const float mu  = rs[0] * (1.f / C_);
    const float var = rss[0] * (1.f / C_) - mu * mu;
    const float inv = rsqrtf(var + 1e-5f);

#pragma unroll
    for (int i = tid; i < C_; i += 256) {
        const float v = (x[i] - mu) * inv * g[i] + b[i];
        Yf[(size_t)blockIdx.x * C_ + i] = v;
        if (Y16) Y16[(size_t)blockIdx.x * C_ + i] = f2bf(v);
    }
}

// ---------------------------------------------------------------------------
// Conv1d(k=3, pad=1) over sequence = 3 tap-shifted GEMMs on WconvT[tap][co][ci]
// Fused + bconv + residual(x) -> y (f32). 32x64 wave tiles.
// ---------------------------------------------------------------------------
__global__ __launch_bounds__(256) void conv_gemm_kernel(
    const unsigned short* __restrict__ X16,   // ln1 bf16 [8192 x 1024]
    const unsigned short* __restrict__ WcT,   // WconvT bf16 [3][1024][1024]
    const float* __restrict__ bias,
    const float* __restrict__ resid,          // ln1 f32
    float* __restrict__ Y) {
    const int lane = threadIdx.x & 31;
    const int wave = blockIdx.x * (blockDim.x >> 5) + (threadIdx.x >> 5);
    const int NT = C_ / 64;
    const int mt = wave / NT, nt = wave % NT;
    const int m0 = mt * 32, n0 = nt * 64;

    v8f acc[2][4];
#pragma unroll
    for (int mh = 0; mh < 2; ++mh)
#pragma unroll
        for (int sub = 0; sub < 4; ++sub) acc[mh][sub] = zero8();

    const int kbA = (lane & 16) ? 8 : 0;

    for (int tap = 0; tap < 3; ++tap) {
        // per-lane shifted rows with zero padding at batch edges
        bool valid[2];
        const unsigned short* arow[2];
#pragma unroll
        for (int mh = 0; mh < 2; ++mh) {
            const int m = m0 + mh * 16 + (lane & 15);
            const int srow = (m & (S_ - 1)) + tap - 1;
            valid[mh] = (unsigned)srow < (unsigned)S_;
            arow[mh] = X16 + (size_t)((m & ~(S_ - 1)) + (valid[mh] ? srow : 0)) * C_;
        }

        for (int k0 = 0; k0 < C_; k0 += 32) {
            FragB16 fa[2];
#pragma unroll
            for (int mh = 0; mh < 2; ++mh)
#pragma unroll
                for (int r = 0; r < 8; ++r) {
                    const int k = k0 + ((r >> 2) << 4) + kbA + 2 * (r & 3);
                    fa[mh].w[r] = valid[mh]
                        ? *reinterpret_cast<const unsigned int*>(arow[mh] + k) : 0u;
                }
            v16bf b[4];
#pragma unroll
            for (int sub = 0; sub < 4; ++sub)
                b[sub] = load_bT_frag(
                    WcT + ((size_t)tap * C_ + n0 + sub * 16) * C_ + k0, C_, lane);
#pragma unroll
            for (int sub = 0; sub < 4; ++sub) {
                acc[0][sub] = wmma_bf16(fa[0].v, b[sub], acc[0][sub]);
                acc[1][sub] = wmma_bf16(fa[1].v, b[sub], acc[1][sub]);
            }
        }
    }

    const int half = (lane & 16) ? 8 : 0;
#pragma unroll
    for (int sub = 0; sub < 4; ++sub) {
        const int n = n0 + sub * 16 + (lane & 15);
        const float bv = bias[n];
#pragma unroll
        for (int mh = 0; mh < 2; ++mh)
#pragma unroll
            for (int r = 0; r < 8; ++r) {
                const size_t idx = (size_t)(m0 + mh * 16 + r + half) * C_ + n;
                Y[idx] = acc[mh][sub][r] + bv + resid[idx];
            }
    }
}

// ---------------------------------------------------------------------------
// Host-side launcher
// ---------------------------------------------------------------------------
extern "C" void kernel_launch(void* const* d_in, const int* in_sizes, int n_in,
                              void* d_out, int out_size, void* d_ws, size_t ws_size,
                              hipStream_t stream) {
    const float* inp   = (const float*)d_in[0];
    const float* Wkqv  = (const float*)d_in[1];
    const float* bkqv  = (const float*)d_in[2];
    const float* Wproj = (const float*)d_in[3];
    const float* bproj = (const float*)d_in[4];
    const float* g1    = (const float*)d_in[5];
    const float* b1    = (const float*)d_in[6];
    const float* Wconv = (const float*)d_in[7];
    const float* bconv = (const float*)d_in[8];
    const float* g2    = (const float*)d_in[9];
    const float* b2    = (const float*)d_in[10];
    float* out = (float*)d_out;

    char* ws = (char*)d_ws;
    size_t off = 0;
    auto alloc = [&](size_t bytes) {
        size_t o = off;
        off = (off + bytes + 255) & ~(size_t)255;
        return o;
    };
    const size_t NTOK = (size_t)BS_ * C_;                 // 8M elements
    unsigned short* inp16   = (unsigned short*)(ws + alloc(NTOK * 2));  // reused as o16
    unsigned short* wkqvT16 = (unsigned short*)(ws + alloc((size_t)C_ * 3 * C_ * 2));
    unsigned short* wprjT16 = (unsigned short*)(ws + alloc((size_t)C_ * C_ * 2));
    unsigned short* wcnvT16 = (unsigned short*)(ws + alloc((size_t)C_ * C_ * 3 * 2));
    unsigned short* q16     = (unsigned short*)(ws + alloc(NTOK * 2));  // reused as ln1 bf16
    unsigned short* k16     = (unsigned short*)(ws + alloc(NTOK * 2));
    unsigned short* vT16    = (unsigned short*)(ws + alloc(NTOK * 2));
    float* xf   = (float*)(ws + alloc(NTOK * 4));                       // reused as y
    float* ln1f = (float*)(ws + alloc(NTOK * 4));

    unsigned short* o16   = inp16;   // inp16 dead after QKV GEMM
    unsigned short* ln116 = q16;     // q16 dead after attention
    float* yf = xf;                  // x dead after LN1

    const int n1 = (int)NTOK;
    const int n2 = C_ * 3 * C_;
    const int n3 = C_ * C_;
    const int n4 = C_ * C_ * 3;
    convert_f32_bf16<<<(n1 + 255) / 256, 256, 0, stream>>>(inp, inp16, n1);
    transpose_f32_bf16<<<(n2 + 255) / 256, 256, 0, stream>>>(Wkqv, wkqvT16, C_, 3 * C_);
    transpose_f32_bf16<<<(n3 + 255) / 256, 256, 0, stream>>>(Wproj, wprjT16, C_, C_);
    conv_w_transpose<<<(n4 + 255) / 256, 256, 0, stream>>>(Wconv, wcnvT16);

    // QKV: (8192/32) x (3072/64) wave tiles / 8 waves per block = 1536 blocks
    qkv_gemm_kernel<<<(BS_ / 32) * (3 * C_ / 64) / 8, 256, 0, stream>>>(
        inp16, wkqvT16, bkqv, q16, k16, vT16);

    // Attention: 8192 query-block waves / 8 = 1024 blocks
    attn_kernel<<<(B_ * H_ * (S_ / 16)) / 8, 256, 0, stream>>>(q16, k16, vT16, o16);

    // Projection + residual: (8192/32) x (1024/64) / 8 = 512 blocks
    proj_gemm_kernel<<<(BS_ / 32) * (C_ / 64) / 8, 256, 0, stream>>>(
        o16, wprjT16, bproj, inp, xf);

    // LayerNorm 1 (f32 + bf16 copy for conv)
    layernorm_kernel<<<BS_, 256, 0, stream>>>(xf, g1, b1, ln1f, ln116);

    // Conv1d as 3 shifted GEMMs + residual
    conv_gemm_kernel<<<(BS_ / 32) * (C_ / 64) / 8, 256, 0, stream>>>(
        ln116, wcnvT16, bconv, ln1f, yf);

    // LayerNorm 2 -> output
    layernorm_kernel<<<BS_, 256, 0, stream>>>(yf, g2, b2, out, nullptr);
}
